// Decoder_22067541967514
// MI455X (gfx1250) — matrix operand order repaired and checked by reading, not compiled, not run on hardware
//
#include <hip/hip_runtime.h>

// ---- problem constants -------------------------------------------------
#define VV    32000
#define EMB   256
#define UU    512
#define BB    32
#define SS    128
#define TT    63
#define U3    1536          // 3*U
#define ROWS  (BB*TT)       // 2016 scan rows

typedef _Float16 f16;
typedef __attribute__((ext_vector_type(16))) _Float16 v16h;
typedef __attribute__((ext_vector_type(8)))  _Float16 v8h;
typedef __attribute__((ext_vector_type(8)))  float    v8f;

#define WMMA(a, b, c) \
  __builtin_amdgcn_wmma_f32_16x16x32_f16(false, (a), false, (b), (short)0, (c), false, false)

// ---- fragment loaders (wave32 WMMA; contiguous k = 16h + s in both A and B) ----------
__device__ __forceinline__ v16h load_a_f16row(const f16* rowptr, int kb) {
  const int lane = threadIdx.x & 31;
  const int kh = lane & 16;                 // 0 or 16
  const v8h lo = *(const v8h*)(rowptr + kb + kh);
  const v8h hi = *(const v8h*)(rowptr + kb + kh + 8);
  v16h a;
#pragma unroll
  for (int e = 0; e < 8; ++e) { a[e] = lo[e]; a[8 + e] = hi[e]; }
  return a;
}
__device__ __forceinline__ v16h load_a_f32row(const float* rowptr, int kb) {
  const int lane = threadIdx.x & 31;
  const int kh = lane & 16;
  v16h a;
#pragma unroll
  for (int e = 0; e < 8; ++e) {
    a[e]     = (f16)rowptr[kb + kh + e];
    a[8 + e] = (f16)rowptr[kb + kh + 8 + e];
  }
  return a;
}
// B 32x16 from transposed f16 weights WT[N][K] (ldT = K): two 16B loads.
__device__ __forceinline__ v16h load_bT_f16(const f16* WT, int ldT, int kb, int col) {
  const int lane = threadIdx.x & 31;
  const int n  = lane & 15;
  const int kh = lane & 16;                 // 0 or 16
  const f16* base = WT + (size_t)(col + n) * ldT + kb + kh;
  const v8h lo = *(const v8h*)(base);
  const v8h hi = *(const v8h*)(base + 8);
  v16h b;
#pragma unroll
  for (int e = 0; e < 8; ++e) { b[e] = lo[e]; b[8 + e] = hi[e]; }
  return b;
}
// B 32x16 from an LDS-staged tile shaped [rows][32] f16 (row = output col).
__device__ __forceinline__ v16h load_b_lds32(const f16* tile, int lcol) {
  const int lane = threadIdx.x & 31;
  const int n  = lane & 15;
  const int kh = lane & 16;
  const f16* base = tile + (size_t)(lcol + n) * 32 + kh;
  const v8h lo = *(const v8h*)(base);
  const v8h hi = *(const v8h*)(base + 8);
  v16h b;
#pragma unroll
  for (int e = 0; e < 8; ++e) { b[e] = lo[e]; b[8 + e] = hi[e]; }
  return b;
}

__device__ __forceinline__ float sigm(float x) { return 1.f / (1.f + expf(-x)); }

// ---- TDM support (6-arg builtin, clang-23/therock-10.0; device pass) ---
#if defined(__HIP_DEVICE_COMPILE__) && __has_builtin(__builtin_amdgcn_tensor_load_to_lds)
#define USE_TDM 1
typedef __attribute__((ext_vector_type(4))) unsigned int tdm_u32x4;
typedef __attribute__((ext_vector_type(8))) int          tdm_i32x8;
typedef __attribute__((ext_vector_type(4))) int          tdm_i32x4;

// Stage a 2-D tile [tile_rows=512][tile_k=32] of f16 (row stride 512 elems)
// from global to LDS.  valid_rows < 512 rows beyond the tensor edge read as
// zero per the D# OOB rule, keeping EXEC uniform for downstream WMMA.
__device__ __forceinline__ void tdm_stage_wo(const f16* gsrc, unsigned lds_off,
                                             int valid_rows) {
  const unsigned long long ga = (unsigned long long)(uintptr_t)gsrc;
  tdm_u32x4 g0 = {};
  g0[0] = 1u;                                             // count=1, user D#
  g0[1] = lds_off;                                        // lds_addr (bytes)
  g0[2] = (unsigned)ga;                                   // global_addr[31:0]
  g0[3] = (unsigned)((ga >> 32) & 0x01ffffffu) | (2u << 30);  // [56:32]|type=2
  tdm_i32x8 g1 = {};
  g1[0] = 1 << 16;                                        // data_size = 2 B
  g1[1] = (UU & 0xffff) << 16;                            // tensor_dim0 = 512
  g1[2] = (valid_rows & 0xffff) << 16;                    // tensor_dim1 lo
  g1[3] = ((valid_rows >> 16) & 0xffff) | (32 << 16);     // dim1 hi | tile_dim0
  g1[4] = 512;                                            // tile_dim1 = 512
  g1[5] = UU;                                             // dim0_stride = 512
  tdm_i32x4 gz4 = {};                                     // groups 2/3 unused (2-D)
  tdm_i32x8 gz8 = {};
  __builtin_amdgcn_tensor_load_to_lds(g0, g1, gz4, gz4, gz8, 0);
}
#endif

// ========================================================================
// K0: generic transpose + fp32->f16 convert: out[C][R] = (f16)in[R][C]
// ========================================================================
__global__ void k_transpose(const float* __restrict__ in, f16* __restrict__ out,
                            int R, int C) {
  const size_t total = (size_t)R * C;
  for (size_t i = (size_t)blockIdx.x * blockDim.x + threadIdx.x; i < total;
       i += (size_t)gridDim.x * blockDim.x) {
    const size_t r = i % R, c = i / R;       // i = c*R + r  (coalesced write)
    out[i] = (f16)in[r * (size_t)C + c];
  }
}

// ========================================================================
// K1: XE[t*B+b][0:1536] = E[x[b,t]] @ K_kernel[0:EMB] + gru_bias[0]
// ========================================================================
__global__ void k_embed(const int* __restrict__ x, const float* __restrict__ E,
                        const f16* __restrict__ KeT, const float* __restrict__ gb,
                        float* __restrict__ XE) {
  const int lane = threadIdx.x & 31;
  const int wv   = threadIdx.x >> 5;
  const int tile = blockIdx.x * 8 + wv;          // 126*96 = 12096 tiles
  const int rt = tile / 96, ct = tile % 96;
  const int r0 = rt * 16, c0 = ct * 16;
  const int n  = lane & 15;
  const int ro = (lane >> 4) << 3;

  const int r  = r0 + (lane & 15);
  const int t  = r >> 5, b = r & 31;             // r = t*B + b
  const float* rowp = E + (size_t)x[b * TT + t] * EMB;

  v8f acc;
  const float bias = gb[c0 + n];
#pragma unroll
  for (int j = 0; j < 8; ++j) acc[j] = bias;

  for (int kc = 0; kc < EMB / 32; ++kc) {
    v16h a = load_a_f32row(rowp, kc * 32);
    v16h w = load_bT_f16(KeT, EMB, kc * 32, c0);
    acc = WMMA(a, w, acc);
  }
#pragma unroll
  for (int j = 0; j < 8; ++j)
    XE[(size_t)(r0 + j + ro) * U3 + c0 + n] = acc[j];
}

// ========================================================================
// K2: keys = memory @ Wk     (4096 x 512, K = 512)
// ========================================================================
__global__ void k_keys(const float* __restrict__ mem, const f16* __restrict__ WkT,
                       float* __restrict__ keys) {
  const int lane = threadIdx.x & 31;
  const int wv   = threadIdx.x >> 5;
  const int tile = blockIdx.x * 8 + wv;          // 256*32 = 8192 tiles
  const int rt = tile / 32, ct = tile % 32;
  const int r0 = rt * 16, c0 = ct * 16;
  const int n  = lane & 15;
  const int ro = (lane >> 4) << 3;

  const float* rowp = mem + (size_t)(r0 + (lane & 15)) * UU;
  v8f acc = {};
  for (int kc = 0; kc < UU / 32; ++kc) {
    v16h a = load_a_f32row(rowp, kc * 32);
    v16h w = load_bT_f16(WkT, UU, kc * 32, c0);
    acc = WMMA(a, w, acc);
  }
#pragma unroll
  for (int j = 0; j < 8; ++j)
    keys[(size_t)(r0 + j + ro) * UU + c0 + n] = acc[j];
}

// ========================================================================
// K3: sequential recurrence; single workgroup, 32 waves, state in LDS.
// ========================================================================
__global__ void __launch_bounds__(1024)
k_decode(const float* __restrict__ enc, const float* __restrict__ mem,
         const float* __restrict__ keys, const float* __restrict__ XE,
         const f16* __restrict__ KaT, const f16* __restrict__ RT,
         const float* __restrict__ gb, const f16* __restrict__ WqT,
         const float* __restrict__ v_att, const f16* __restrict__ WaT,
         f16* __restrict__ ATTNg) {
  extern __shared__ char smem[];
  f16*   sh_h    = (f16*)smem;                       // [32][512]
  f16*   sh_attn = sh_h    + BB * UU;                // [32][512]
  f16*   sh_pq   = sh_attn + BB * UU;                // [32][512]
  f16*   sh_ctx  = sh_pq   + BB * UU;                // [32][512]
  float* sh_sc   = (float*)(sh_ctx + BB * UU);       // [32][128]
  float* sh_v    = sh_sc  + BB * SS;                 // [512]
  float* sh_gb1  = sh_v   + UU;                      // [1536]

  const int tid  = threadIdx.x;
  const int lane = tid & 31;
  const int wv   = tid >> 5;          // 32 waves <-> 32 column tiles of U
  const int n    = lane & 15;
  const int ro   = (lane >> 4) << 3;
  const int c0   = wv * 16;

  for (int i = tid; i < BB * UU; i += 1024) {
    sh_h[i]    = (f16)enc[i];
    sh_attn[i] = (f16)0.f;
  }
  for (int i = tid; i < UU; i += 1024) sh_v[i] = v_att[i];
  for (int i = tid; i < U3; i += 1024) sh_gb1[i] = gb[U3 + i];
  __syncthreads();

  for (int t = 0; t < TT; ++t) {
    // ---------- Phase 1: gates; acc init folds XE[t] (incl. bias0) + bias1
    const float* xe_t = XE + (size_t)t * BB * U3;
    v8f accZ[2], accR[2], accXH[2], accHHR[2];
    {
      const float gz = sh_gb1[c0 + n];
      const float gr = sh_gb1[UU + c0 + n];
      const float gh = sh_gb1[2 * UU + c0 + n];
#pragma unroll
      for (int rb = 0; rb < 2; ++rb)
#pragma unroll
        for (int j = 0; j < 8; ++j) {
          const size_t row = rb * 16 + j + ro;
          accZ[rb][j]   = xe_t[row * U3 + c0 + n] + gz;
          accR[rb][j]   = xe_t[row * U3 + UU + c0 + n] + gr;
          accXH[rb][j]  = xe_t[row * U3 + 2 * UU + c0 + n];
          accHHR[rb][j] = gh;
        }
    }
    for (int kc = 0; kc < UU / 32; ++kc) {
      const int kb = kc * 32;
      v16h aA0 = load_a_f16row(sh_attn + (size_t)(0  + (lane & 15)) * UU, kb);
      v16h aA1 = load_a_f16row(sh_attn + (size_t)(16 + (lane & 15)) * UU, kb);
      v16h aH0 = load_a_f16row(sh_h    + (size_t)(0  + (lane & 15)) * UU, kb);
      v16h aH1 = load_a_f16row(sh_h    + (size_t)(16 + (lane & 15)) * UU, kb);
      v16h bKz = load_bT_f16(KaT, UU, kb, c0);
      v16h bKr = load_bT_f16(KaT, UU, kb, UU + c0);
      v16h bKh = load_bT_f16(KaT, UU, kb, 2 * UU + c0);
      v16h bRz = load_bT_f16(RT,  UU, kb, c0);
      v16h bRr = load_bT_f16(RT,  UU, kb, UU + c0);
      v16h bRh = load_bT_f16(RT,  UU, kb, 2 * UU + c0);
      accZ[0]   = WMMA(aA0, bKz, accZ[0]);   accZ[0]   = WMMA(aH0, bRz, accZ[0]);
      accZ[1]   = WMMA(aA1, bKz, accZ[1]);   accZ[1]   = WMMA(aH1, bRz, accZ[1]);
      accR[0]   = WMMA(aA0, bKr, accR[0]);   accR[0]   = WMMA(aH0, bRr, accR[0]);
      accR[1]   = WMMA(aA1, bKr, accR[1]);   accR[1]   = WMMA(aH1, bRr, accR[1]);
      accXH[0]  = WMMA(aA0, bKh, accXH[0]);
      accXH[1]  = WMMA(aA1, bKh, accXH[1]);
      accHHR[0] = WMMA(aH0, bRh, accHHR[0]);
      accHHR[1] = WMMA(aH1, bRh, accHHR[1]);
    }
    __syncthreads();                       // all reads of sh_h/sh_attn done
#pragma unroll
    for (int rb = 0; rb < 2; ++rb)
#pragma unroll
      for (int j = 0; j < 8; ++j) {
        const int row = rb * 16 + j + ro, col = c0 + n;
        const float z  = sigm(accZ[rb][j]);
        const float r  = sigm(accR[rb][j]);
        const float hh = tanhf(accXH[rb][j] + r * accHHR[rb][j]);
        const float ho = (float)sh_h[(size_t)row * UU + col];
        sh_h[(size_t)row * UU + col] = (f16)(z * ho + (1.f - z) * hh);
      }
    __syncthreads();

    // ---------- Phase 2: pq = h_new @ Wq
    {
      v8f p0 = {}, p1 = {};
      for (int kc = 0; kc < UU / 32; ++kc) {
        const int kb = kc * 32;
        v16h aH0 = load_a_f16row(sh_h + (size_t)(0  + (lane & 15)) * UU, kb);
        v16h aH1 = load_a_f16row(sh_h + (size_t)(16 + (lane & 15)) * UU, kb);
        v16h bq  = load_bT_f16(WqT, UU, kb, c0);
        p0 = WMMA(aH0, bq, p0);
        p1 = WMMA(aH1, bq, p1);
      }
#pragma unroll
      for (int j = 0; j < 8; ++j) {
        sh_pq[(size_t)(j + ro) * UU + c0 + n]      = (f16)p0[j];
        sh_pq[(size_t)(16 + j + ro) * UU + c0 + n] = (f16)p1[j];
      }
    }
    __syncthreads();

    // ---------- Phase 3: scores, softmax, context
    for (int p = tid; p < BB * SS; p += 1024) {
      const int b = p >> 7, s = p & (SS - 1);
      const float* kr = keys + ((size_t)b * SS + s) * UU;
      const f16*   pr = sh_pq + (size_t)b * UU;
      float acc = 0.f;
#pragma unroll 4
      for (int u = 0; u < UU; ++u)
        acc += sh_v[u] * tanhf(kr[u] + (float)pr[u]);
      sh_sc[p] = acc;
    }
    __syncthreads();
    {                                       // softmax: wave wv handles b = wv
      const int b = wv;
      float vals[4], m = -3.4e38f;
#pragma unroll
      for (int i = 0; i < 4; ++i) {
        vals[i] = sh_sc[b * SS + lane + 32 * i];
        m = fmaxf(m, vals[i]);
      }
#pragma unroll
      for (int off = 16; off > 0; off >>= 1) m = fmaxf(m, __shfl_xor(m, off));
      float sum = 0.f;
#pragma unroll
      for (int i = 0; i < 4; ++i) { vals[i] = expf(vals[i] - m); sum += vals[i]; }
#pragma unroll
      for (int off = 16; off > 0; off >>= 1) sum += __shfl_xor(sum, off);
      const float inv = 1.f / sum;
#pragma unroll
      for (int i = 0; i < 4; ++i) sh_sc[b * SS + lane + 32 * i] = vals[i] * inv;
    }
    __syncthreads();
    for (int p = tid; p < BB * UU; p += 1024) {
      const int b = p >> 9, u = p & (UU - 1);
      const float* mr = mem + (size_t)b * SS * UU + u;
      float acc = 0.f;
#pragma unroll 4
      for (int s = 0; s < SS; ++s)
        acc += sh_sc[b * SS + s] * mr[(size_t)s * UU];
      sh_ctx[p] = (f16)acc;
    }
    __syncthreads();

    // ---------- Phase 4: attn_new = [h_new | context] @ Wa
    {
      v8f a0 = {}, a1 = {};
      for (int kc = 0; kc < 2 * UU / 32; ++kc) {
        const f16* src = (kc < 16) ? sh_h : sh_ctx;
        const int  kb  = (kc < 16) ? kc * 32 : (kc - 16) * 32;
        v16h f0 = load_a_f16row(src + (size_t)(0  + (lane & 15)) * UU, kb);
        v16h f1 = load_a_f16row(src + (size_t)(16 + (lane & 15)) * UU, kb);
        v16h bw = load_bT_f16(WaT, 2 * UU, kc * 32, c0);
        a0 = WMMA(f0, bw, a0);
        a1 = WMMA(f1, bw, a1);
      }
#pragma unroll
      for (int j = 0; j < 8; ++j) {
        const int r0c = j + ro, r1c = 16 + j + ro, col = c0 + n;
        sh_attn[(size_t)r0c * UU + col] = (f16)a0[j];
        sh_attn[(size_t)r1c * UU + col] = (f16)a1[j];
        ATTNg[((size_t)t * BB + r0c) * UU + col] = (f16)a0[j];
        ATTNg[((size_t)t * BB + r1c) * UU + col] = (f16)a1[j];
      }
    }
    __syncthreads();
  }
}

// ========================================================================
// K4: logits = ATTN @ Wo + bo   ([2016 x 512] x [512 x 32000])
// 8 waves/block, 32 rows x 512 cols per block; wave owns 32x64 (8 accs).
// TDM path: wave 0 DMA-stages each [512 x 32] f16 WoT tile into LDS
// (double-buffered, tensor_dim1 handles the ragged vocab edge with zeros),
// s_wait_tensorcnt + barrier hands it to all waves.
// ========================================================================
__global__ void __launch_bounds__(256)
k_logits(const f16* __restrict__ A, const f16* __restrict__ WoT,
         const float* __restrict__ bo, float* __restrict__ out) {
  extern __shared__ char lsm[];
  const int lane = threadIdx.x & 31;
  const int wv   = threadIdx.x >> 5;
  const int n    = lane & 15;
  const int ro   = (lane >> 4) << 3;
  const int rowbase = blockIdx.x * 32;                 // 63 * 32 = 2016
  const int cblock  = blockIdx.y * 512;                // 63 * 512 >= 32000
  const int c0 = cblock + wv * 64;

  int ccol[4];
#pragma unroll
  for (int i = 0; i < 4; ++i) {
    const int c = c0 + 16 * i;
    ccol[i] = (c < VV) ? c : (VV - 16);                // clamp (bo init/stores)
  }

  const f16* rowp0 = A + (size_t)(rowbase +      (lane & 15)) * UU;
  const f16* rowp1 = A + (size_t)(rowbase + 16 + (lane & 15)) * UU;

  v8f acc[2][4];
#pragma unroll
  for (int i = 0; i < 4; ++i) {
    const float bv = bo[ccol[i] + n];
#pragma unroll
    for (int j = 0; j < 8; ++j) { acc[0][i][j] = bv; acc[1][i][j] = bv; }
  }

#ifdef USE_TDM
  f16* shW0 = (f16*)lsm;                               // [512][32] f16 = 32 KB
  f16* shW1 = shW0 + 512 * 32;                         // double buffer
  const unsigned off0 = (unsigned)(uintptr_t)shW0;
  const unsigned off1 = (unsigned)(uintptr_t)shW1;
  const int valid_rows = (VV - cblock < 512) ? (VV - cblock) : 512;
  const f16* gtile = WoT + (size_t)cblock * UU;

  if (wv == 0) tdm_stage_wo(gtile, off0, valid_rows);  // prologue: chunk 0
  for (int kc = 0; kc < UU / 32; ++kc) {
    if (wv == 0) __builtin_amdgcn_s_wait_tensorcnt(0); // current tile landed
    __syncthreads();                                   // hand tile to all waves
    if (kc + 1 < UU / 32 && wv == 0)                   // overlap next DMA
      tdm_stage_wo(gtile + (kc + 1) * 32, (kc & 1) ? off0 : off1, valid_rows);
    const f16* cw = (kc & 1) ? shW1 : shW0;
    const int kb = kc * 32;
    v16h a0 = load_a_f16row(rowp0, kb);
    v16h a1 = load_a_f16row(rowp1, kb);
#pragma unroll
    for (int i = 0; i < 4; ++i) {
      v16h w = load_b_lds32(cw, wv * 64 + 16 * i);     // local cols, EXEC full
      acc[0][i] = WMMA(a0, w, acc[0][i]);
      acc[1][i] = WMMA(a1, w, acc[1][i]);
    }
  }
#else
  for (int kc = 0; kc < UU / 32; ++kc) {
    const int kb = kc * 32;
    v16h a0 = load_a_f16row(rowp0, kb);
    v16h a1 = load_a_f16row(rowp1, kb);
#pragma unroll
    for (int i = 0; i < 4; ++i) {
      v16h w = load_bT_f16(WoT, UU, kb, ccol[i]);
      acc[0][i] = WMMA(a0, w, acc[0][i]);
      acc[1][i] = WMMA(a1, w, acc[1][i]);
    }
  }
#endif

#pragma unroll
  for (int i = 0; i < 4; ++i) {
    if (c0 + 16 * i >= VV) continue;                   // store guard only
#pragma unroll
    for (int rt = 0; rt < 2; ++rt)
#pragma unroll
      for (int j = 0; j < 8; ++j) {
        const int row = rowbase + rt * 16 + j + ro;    // row = t*B + b
        const int t = row >> 5, b = row & 31;
        out[((size_t)b * TT + t) * VV + ccol[i] + n] = acc[rt][i][j];
      }
  }
}

// ========================================================================
extern "C" void kernel_launch(void* const* d_in, const int* in_sizes, int n_in,
                              void* d_out, int out_size, void* d_ws, size_t ws_size,
                              hipStream_t stream) {
  const int*   x    = (const int*)  d_in[0];
  const float* enc  = (const float*)d_in[1];
  const float* mem  = (const float*)d_in[2];
  const float* E    = (const float*)d_in[3];
  const float* Kk   = (const float*)d_in[4];
  const float* Rk   = (const float*)d_in[5];
  const float* gb   = (const float*)d_in[6];
  const float* Wq   = (const float*)d_in[7];
  const float* Wk   = (const float*)d_in[8];
  const float* vatt = (const float*)d_in[9];
  const float* Wa   = (const float*)d_in[10];
  const float* Wo   = (const float*)d_in[11];
  const float* bo   = (const float*)d_in[12];
  float* out = (float*)d_out;

  // ---- workspace carve-out (~62 MB) ----
  float* XE    = (float*)d_ws;                         // [2016][1536] f32
  float* keys  = XE + (size_t)ROWS * U3;               // [4096][512]  f32
  f16*   ATTNg = (f16*)(keys + (size_t)BB * SS * UU);  // [63][32][512] f16
  f16*   WoT   = ATTNg + (size_t)ROWS * UU;            // [32000][512] f16
  f16*   KeT   = WoT + (size_t)VV * UU;                // [1536][256]  f16
  f16*   KaT   = KeT + (size_t)U3 * EMB;               // [1536][512]  f16
  f16*   RT    = KaT + (size_t)U3 * UU;                // [1536][512]  f16
  f16*   WqT   = RT  + (size_t)U3 * UU;                // [512][512]   f16
  f16*   WkT   = WqT + (size_t)UU * UU;                // [512][512]   f16
  f16*   WaT   = WkT + (size_t)UU * UU;                // [512][1024]  f16

  // ---- K0: one-shot weight transpose + f16 convert (amortized 63..126x)
  k_transpose<<<4096, 256, 0, stream>>>(Wo, WoT, UU, VV);
  k_transpose<<<512,  256, 0, stream>>>(Kk, KeT, EMB, U3);
  k_transpose<<<512,  256, 0, stream>>>(Kk + (size_t)EMB * U3, KaT, UU, U3);
  k_transpose<<<512,  256, 0, stream>>>(Rk, RT,  UU, U3);
  k_transpose<<<256,  256, 0, stream>>>(Wq, WqT, UU, UU);
  k_transpose<<<256,  256, 0, stream>>>(Wk, WkT, UU, UU);
  k_transpose<<<256,  256, 0, stream>>>(Wa, WaT, 2 * UU, UU);

  // ---- K1: embedding half of GRU input matmul (126*96 tiles / 8 waves)
  k_embed<<<1512, 256, 0, stream>>>(x, E, KeT, gb, XE);
  // ---- K2: attention keys (256*32 tiles / 8 waves)
  k_keys<<<1024, 256, 0, stream>>>(mem, WkT, keys);

  // ---- K3: persistent single-workgroup scan, 152 KB dynamic LDS
  const size_t smem3 = (size_t)4 * BB * UU * sizeof(f16)
                     + (size_t)(BB * SS + UU + U3) * sizeof(float);
  k_decode<<<1, 1024, smem3, stream>>>(enc, mem, keys, XE, KaT, RT, gb,
                                       WqT, vatt, WaT, ATTNg);

  // ---- K4: vocab projection, 63x63 blocks; 64 KB LDS for TDM double buffer
  k_logits<<<dim3(63, 63), 256, 64 * 1024, stream>>>(ATTNg, WoT, bo, out);
}
